// LSTMStacked_54296976556204
// MI455X (gfx1250) — matrix operand order repaired and checked by reading, not compiled, not run on hardware
//
#include <hip/hip_runtime.h>
#include <cstdint>
#include <cstddef>

// ---------------------------------------------------------------------------
// CDNA5 (gfx1250) wave32 WMMA types
// ---------------------------------------------------------------------------
typedef __bf16 v16bf __attribute__((ext_vector_type(16)));
typedef float  v8f   __attribute__((ext_vector_type(8)));

struct __align__(16) Chunk16 { unsigned int w[4]; };  // 16 bytes = 8 bf16
union FragB16 {
  v16bf   v;
  Chunk16 q[2];
};

__device__ __forceinline__ unsigned short f2bf(float f) {
  // round-to-nearest-even fp32 -> bf16
  unsigned int u = __float_as_uint(f);
  u += 0x7FFFu + ((u >> 16) & 1u);
  return (unsigned short)(u >> 16);
}

__device__ __forceinline__ float fast_sigmoid(float x) {
  return 1.0f / (1.0f + __expf(-x));
}

// ---------------------------------------------------------------------------
// Phase 1: data preparation
// ---------------------------------------------------------------------------

// x [B=32, T=2048, D=64] fp32  ->  xb [(t*32+b)*64 + k] bf16  ([T,B,D] order)
__global__ void prep_x_kernel(const float* __restrict__ x,
                              unsigned short* __restrict__ xb) {
  int idx = blockIdx.x * blockDim.x + threadIdx.x;   // exact grid
  int k = idx & 63;
  int r = idx >> 6;
  int b = r & 31;
  int t = r >> 5;
  xb[idx] = f2bf(x[((size_t)b * 2048 + t) * 64 + k]);
}

// Swizzle torch weight W [N,K] (row-major fp32) into WMMA B-fragment order:
// tile (nt,kt): 32 lanes x 16 halves contiguous; lane l holds column
// N = nt*16 + (l&15), halves h -> K = kt*32 + h + 16*(l>>4).
__global__ void swizzle_b_kernel(const float* __restrict__ W,
                                 unsigned short* __restrict__ out,
                                 int N, int K) {
  int idx = blockIdx.x * blockDim.x + threadIdx.x;
  if (idx >= N * K) return;
  int h    = idx & 15;
  int lane = (idx >> 4) & 31;
  int tile = idx >> 9;            // nt*KT + kt
  int KT = K >> 5;
  int nt = tile / KT;
  int kt = tile - nt * KT;
  int n = nt * 16 + (lane & 15);
  int k = kt * 32 + h + 16 * (lane >> 4);
  out[idx] = f2bf(W[(size_t)n * K + k]);
}

__global__ void add_bias_kernel(const float* __restrict__ a,
                                const float* __restrict__ b,
                                float* __restrict__ o, int n) {
  int i = blockIdx.x * blockDim.x + threadIdx.x;
  if (i < n) o[i] = a[i] + b[i];
}

// ---------------------------------------------------------------------------
// Generic bf16 WMMA GEMM:  out[M,N] = act(A[M,K] * Bsw^T + bias)
// Each wave computes one 16 x (16*NT) strip.  A: row-major bf16, rows = (t,b).
// ---------------------------------------------------------------------------
template <int NT>
__global__ __launch_bounds__(256) void gemm_bf16_kernel(
    const unsigned short* __restrict__ A,
    const unsigned short* __restrict__ Bsw,
    const float* __restrict__ bias,
    int M, int N, int K, int relu,
    float* __restrict__ outF,
    unsigned short* __restrict__ outB) {
  const int lane   = threadIdx.x & 31;
  const int waveId = blockIdx.x * (blockDim.x >> 5) + (threadIdx.x >> 5);
  const int wavesPerRow = N / (16 * NT);
  const int mt = waveId / wavesPerRow;
  const int ng = waveId - mt * wavesPerRow;
  if (mt * 16 >= M) return;                 // wave-uniform exit (EXEC stays full)
  const int KT = K >> 5;
  const int g  = lane >> 4;
  const int m  = mt * 16 + (lane & 15);

  const v8f zero = {0.f, 0.f, 0.f, 0.f, 0.f, 0.f, 0.f, 0.f};
  v8f acc[NT];
#pragma unroll
  for (int j = 0; j < NT; ++j) acc[j] = zero;

  for (int k = 0; k < KT; ++k) {
    FragB16 a;
    const unsigned short* ar = A + (size_t)m * K + k * 32;
    a.q[0] = *reinterpret_cast<const Chunk16*>(ar + 16 * g);
    a.q[1] = *reinterpret_cast<const Chunk16*>(ar + 16 * g + 8);
#pragma unroll
    for (int j = 0; j < NT; ++j) {
      const int nt = ng * NT + j;
      FragB16 b;
      const unsigned short* bp = Bsw + (((size_t)nt * KT + k) * 32 + lane) * 16;
      b.q[0] = *reinterpret_cast<const Chunk16*>(bp);
      b.q[1] = *reinterpret_cast<const Chunk16*>(bp + 8);
      acc[j] = __builtin_amdgcn_wmma_f32_16x16x32_bf16(
          false, a.v, false, b.v, (short)0, acc[j], false, false);
    }
  }

#pragma unroll
  for (int j = 0; j < NT; ++j) {
    const int n  = (ng * NT + j) * 16 + (lane & 15);
    const float bv = bias ? bias[n] : 0.0f;
#pragma unroll
    for (int r = 0; r < 8; ++r) {
      float v = acc[j][r] + bv;
      if (relu) v = fmaxf(v, 0.0f);
      const int row = mt * 16 + r + 8 * g;   // D layout: lanes>=16 hold M=8+r
      const size_t o = (size_t)row * N + n;
      if (outF) outF[o] = v;
      if (outB) outB[o] = f2bf(v);
    }
  }
}

// ---------------------------------------------------------------------------
// Persistent single-WGP LSTM scan.  512 threads = 16 waves; wave w owns
// h-columns [16w, 16w+16).  c stays in VGPRs for all T steps; h lives in LDS.
// Per step, per wave: 4 gates x 2 M-tiles x 8 K-steps = 64 WMMAs.
// __launch_bounds__(512, 1): single resident block -> give each wave the
// maximum VGPR budget (prevents scratch spills in the serial loop).
// ---------------------------------------------------------------------------
__global__ __launch_bounds__(512, 1) void lstm_scan_kernel(
    const float* __restrict__ xg,              // [(t*32+b), 1024] fp32 gate preacts
    const unsigned short* __restrict__ WhhSw,  // swizzled [64nt][8kt][32][16] bf16
    const float* __restrict__ h0,              // [32,256] fp32
    const float* __restrict__ c0,              // [32,256] fp32
    unsigned short* __restrict__ y,            // [(t*32+b), 256] bf16
    float* __restrict__ hn, float* __restrict__ cn,  // [32,256] fp32 each
    int T) {
  __shared__ __align__(16) unsigned short hbuf[32 * 256];

  const int tid  = threadIdx.x;
  const int wave = tid >> 5;
  const int lane = tid & 31;
  const int g    = lane >> 4;
  const int nc   = wave * 16 + (lane & 15);    // owned h-column

  for (int i = tid; i < 32 * 256; i += 512) hbuf[i] = f2bf(h0[i]);

  const v8f zero = {0.f, 0.f, 0.f, 0.f, 0.f, 0.f, 0.f, 0.f};
  v8f cacc[2];
#pragma unroll
  for (int mt = 0; mt < 2; ++mt) {
    cacc[mt] = zero;
#pragma unroll
    for (int r = 0; r < 8; ++r)
      cacc[mt][r] = c0[(mt * 16 + r + 8 * g) * 256 + nc];
  }
  __syncthreads();

  for (int t = 0; t < T; ++t) {
    const float* xrow = xg + (size_t)t * 32 * 1024;
    // Prefetch this step's 128KB of gate pre-activations (one 128B line each).
    for (int i = tid; i < 1024; i += 512)
      __builtin_prefetch(xrow + i * 32, 0, 0);

    v8f acc[4][2];
#pragma unroll
    for (int gi = 0; gi < 4; ++gi) { acc[gi][0] = zero; acc[gi][1] = zero; }

    // Limited unroll: enough outstanding loads to cover L2 latency without
    // blowing the VGPR budget (full unroll caused scratch spills).
#pragma unroll 2
    for (int k = 0; k < 8; ++k) {
      FragB16 a[2];
#pragma unroll
      for (int mt = 0; mt < 2; ++mt) {
        const unsigned short* ar = &hbuf[(mt * 16 + (lane & 15)) * 256 + k * 32];
        a[mt].q[0] = *reinterpret_cast<const Chunk16*>(ar + 16 * g);
        a[mt].q[1] = *reinterpret_cast<const Chunk16*>(ar + 16 * g + 8);
      }
#pragma unroll
      for (int gi = 0; gi < 4; ++gi) {
        const int nt = gi * 16 + wave;         // gate gi, columns nc
        FragB16 b;
        const unsigned short* bp = WhhSw + (((size_t)nt * 8 + k) * 32 + lane) * 16;
        b.q[0] = *reinterpret_cast<const Chunk16*>(bp);
        b.q[1] = *reinterpret_cast<const Chunk16*>(bp + 8);
#pragma unroll
        for (int mt = 0; mt < 2; ++mt)
          acc[gi][mt] = __builtin_amdgcn_wmma_f32_16x16x32_bf16(
              false, a[mt].v, false, b.v, (short)0, acc[gi][mt], false, false);
      }
    }
    __syncthreads();   // all waves finished reading h(t-1) from LDS

#pragma unroll
    for (int mt = 0; mt < 2; ++mt) {
#pragma unroll
      for (int r = 0; r < 8; ++r) {
        const int b = mt * 16 + r + 8 * g;     // batch row
        const float* xb = xrow + (size_t)b * 1024 + nc;
        float iv = fast_sigmoid(acc[0][mt][r] + xb[0]);
        float fv = fast_sigmoid(acc[1][mt][r] + xb[256]);
        float gv = tanhf(acc[2][mt][r] + xb[512]);
        float ov = fast_sigmoid(acc[3][mt][r] + xb[768]);
        float cv = fv * cacc[mt][r] + iv * gv;
        cacc[mt][r] = cv;
        float hv = ov * tanhf(cv);
        unsigned short hb = f2bf(hv);
        hbuf[b * 256 + nc] = hb;
        y[((size_t)t * 32 + b) * 256 + nc] = hb;
        if (t == T - 1) {
          hn[b * 256 + nc] = hv;
          cn[b * 256 + nc] = cv;
        }
      }
    }
    __syncthreads();   // h(t) visible in LDS before next step's WMMAs
  }
}

// outTmp [(t*32+b)*32 + o] fp32 -> d_out[(b*T + t)*32 + o]
__global__ void reorder_out_kernel(const float* __restrict__ tmp,
                                   float* __restrict__ out, int T) {
  int idx = blockIdx.x * blockDim.x + threadIdx.x;   // exact grid: 32*T*32
  int o = idx & 31;
  int r = idx >> 5;
  int b = r & 31;
  int t = r >> 5;
  out[((size_t)b * T + t) * 32 + o] = tmp[idx];
}

// ---------------------------------------------------------------------------
// Host launch
// ---------------------------------------------------------------------------
extern "C" void kernel_launch(void* const* d_in, const int* in_sizes, int n_in,
                              void* d_out, int out_size, void* d_ws, size_t ws_size,
                              hipStream_t stream) {
  (void)in_sizes; (void)n_in; (void)out_size; (void)ws_size;

  const float* x    = (const float*)d_in[0];
  const float* h0   = (const float*)d_in[1];
  const float* c0   = (const float*)d_in[2];
  const float* Wih0 = (const float*)d_in[3];
  const float* Whh0 = (const float*)d_in[4];
  const float* bih0 = (const float*)d_in[5];
  const float* bhh0 = (const float*)d_in[6];
  const float* Wih1 = (const float*)d_in[7];
  const float* Whh1 = (const float*)d_in[8];
  const float* bih1 = (const float*)d_in[9];
  const float* bhh1 = (const float*)d_in[10];
  const float* Wd0  = (const float*)d_in[11];
  const float* bd0  = (const float*)d_in[12];
  const float* Wd1  = (const float*)d_in[13];
  const float* bd1  = (const float*)d_in[14];
  const float* Wout = (const float*)d_in[15];
  const float* bout = (const float*)d_in[16];

  const int B = 32, T = 2048, D = 64, H = 256, G = 4 * H, O = 32;
  const int M = B * T;   // 65536 rows in (t,b) order

  char* ws = (char*)d_ws;
  size_t off = 0;
  auto carve = [&](size_t bytes) {
    size_t o = off;
    off = (off + bytes + 255) & ~(size_t)255;
    return o;
  };
  unsigned short* xb     = (unsigned short*)(ws + carve((size_t)M * D * 2));
  unsigned short* wih0sw = (unsigned short*)(ws + carve((size_t)G * D * 2));
  unsigned short* whh0sw = (unsigned short*)(ws + carve((size_t)G * H * 2));
  unsigned short* wih1sw = (unsigned short*)(ws + carve((size_t)G * H * 2));
  unsigned short* whh1sw = (unsigned short*)(ws + carve((size_t)G * H * 2));
  unsigned short* wd0sw  = (unsigned short*)(ws + carve((size_t)H * H * 2));
  unsigned short* wd1sw  = (unsigned short*)(ws + carve((size_t)H * H * 2));
  unsigned short* woutsw = (unsigned short*)(ws + carve((size_t)O * H * 2));
  float* bias0 = (float*)(ws + carve((size_t)G * 4));
  float* bias1 = (float*)(ws + carve((size_t)G * 4));
  unsigned short* y0 = (unsigned short*)(ws + carve((size_t)M * H * 2));
  unsigned short* y1 = (unsigned short*)(ws + carve((size_t)M * H * 2));
  char* xgArea = ws + carve((size_t)M * G * 4);          // 256 MB
  float* xg = (float*)xgArea;
  // Head intermediates reuse the xg region (xg dead after second scan).
  unsigned short* dbuf0 = (unsigned short*)xgArea;
  unsigned short* dbuf1 = (unsigned short*)(xgArea + (size_t)M * H * 2);
  float* outTmp         = (float*)(xgArea + 2 * (size_t)M * H * 2);

  float* outp = (float*)d_out;
  float* hn = outp + (size_t)B * T * O;   // h_n [2,32,256]
  float* cn = hn + (size_t)2 * B * H;     // c_n [2,32,256]

  // ---- phase 1: convert + swizzle ----
  prep_x_kernel<<<(M * D) / 256, 256, 0, stream>>>(x, xb);
  auto swz = [&](const float* W, unsigned short* o, int n, int k) {
    int tot = n * k;
    swizzle_b_kernel<<<(tot + 255) / 256, 256, 0, stream>>>(W, o, n, k);
  };
  swz(Wih0, wih0sw, G, D);
  swz(Whh0, whh0sw, G, H);
  swz(Wih1, wih1sw, G, H);
  swz(Whh1, whh1sw, G, H);
  swz(Wd0,  wd0sw,  H, H);
  swz(Wd1,  wd1sw,  H, H);
  swz(Wout, woutsw, O, H);
  add_bias_kernel<<<4, 256, 0, stream>>>(bih0, bhh0, bias0, G);
  add_bias_kernel<<<4, 256, 0, stream>>>(bih1, bhh1, bias1, G);

  auto gemm4 = [&](const unsigned short* A, const unsigned short* Bw,
                   const float* bias, int n, int k, int relu,
                   float* oF, unsigned short* oB) {
    int waves = (M / 16) * (n / 64);
    gemm_bf16_kernel<4><<<waves / 8, 256, 0, stream>>>(A, Bw, bias, M, n, k,
                                                       relu, oF, oB);
  };

  // ---- layer 0 ----
  gemm4(xb, wih0sw, bias0, G, D, 0, xg, nullptr);
  lstm_scan_kernel<<<1, 512, 0, stream>>>(xg, whh0sw, h0, c0, y0, hn, cn, T);

  // ---- layer 1 ----
  gemm4(y0, wih1sw, bias1, G, H, 0, xg, nullptr);
  lstm_scan_kernel<<<1, 512, 0, stream>>>(xg, whh1sw, h0 + B * H, c0 + B * H,
                                          y1, hn + B * H, cn + B * H, T);

  // ---- dense head ----
  gemm4(y1, wd0sw, bd0, H, H, 1, nullptr, dbuf0);
  gemm4(dbuf0, wd1sw, bd1, H, H, 1, nullptr, dbuf1);
  {
    int waves = (M / 16) * (O / 32);   // NT=2 -> 32 columns per wave
    gemm_bf16_kernel<2><<<waves / 8, 256, 0, stream>>>(dbuf1, woutsw, bout, M, O,
                                                       H, 0, outTmp, nullptr);
  }
  reorder_out_kernel<<<(M * O) / 256, 256, 0, stream>>>(outTmp, outp, T);
}
